// SpeechClassifierDP_71511205478932
// MI455X (gfx1250) — matrix operand order repaired and checked by reading, not compiled, not run on hardware
//
#include <hip/hip_runtime.h>
#include <cstdint>
#include <cstddef>

// ============================================================================
// SpeechClassifierDP — interval-bound-propagation classifier on MI455X
// (gfx1250, wave32, WMMA bf16 -> f32 accumulate).
//
// Pipeline:
//   frame -> ilin(mtx1) -> interval-square -> ilin(mtx2)+1e-10 -> log
//         -> ilin(lin1)+b -> relu -> 320x { LSTM cell L0, LSTM cell L1 }
//         -> ilin(lin2)+b -> d_out (2,128,10)
//
// Every interval-linear is a dual GEMM: P = center @ W, Q = radius @ |W|,
// l = P - Q + b, u = P + Q + b.  Weights are packed once into the hardware
// B-fragment layout (32x16 block = 1024B, 32B/lane).  The sequential LSTM
// gate GEMM (the 98%-of-FLOPs part) uses a 2x2 super-tile per wave: 8 WMMAs
// per 8 fragment loads per K-block.
// ============================================================================

typedef __bf16 bf16_t;
typedef __attribute__((ext_vector_type(16))) __bf16 v16bf;
typedef __attribute__((ext_vector_type(8)))  __bf16 v8bf;
typedef __attribute__((ext_vector_type(8)))  float  v8f;

#define DEV static __device__ __forceinline__

DEV bf16_t f2bf(float f) {
  union { float f; uint32_t u; } v; v.f = f;
  uint32_t r = (v.u + 0x7FFFu + ((v.u >> 16) & 1u)) >> 16;  // round-nearest-even
  uint16_t h = (uint16_t)r;
  return __builtin_bit_cast(bf16_t, h);
}

DEV float sigf(float x) { return 1.0f / (1.0f + __expf(-x)); }

DEV void imul(float al, float au, float bl, float bu, float& lo, float& hi) {
  float p0 = al * bl, p1 = al * bu, p2 = au * bl, p3 = au * bu;
  lo = fminf(fminf(p0, p1), fminf(p2, p3));
  hi = fmaxf(fmaxf(p0, p1), fmaxf(p2, p3));
}

// --- fragment loaders ------------------------------------------------------
// A (16x32 bf16, row-major, stride lda): lanes 0-15 row = lane, K 0..7 in
// v0..3 and K 16..23 in v4..7; lanes 16-31 same rows, K +8.
// Per-lane base pointer precomputed once; K advances by constant offsets so
// the backend folds them into the 24-bit instruction offset.
DEV const bf16_t* a_base(const bf16_t* __restrict__ A, size_t lda, int row0, int lane) {
  return A + (size_t)(row0 + (lane & 15)) * lda + ((lane >> 4) << 3);
}
DEV v16bf load_a_at(const bf16_t* __restrict__ base, int k0) {
  v8bf lo = *(const v8bf*)(base + k0);
  v8bf hi = *(const v8bf*)(base + k0 + 16);
  v16bf a;
#pragma unroll
  for (int i = 0; i < 8; ++i) { a[i] = lo[i]; a[i + 8] = hi[i]; }
  return a;
}
// B from pre-packed buffer: block = 512 bf16 (1024B), lane reads 16 bf16.
DEV const bf16_t* b_base(const bf16_t* __restrict__ Bpk, int blockStart, int lane) {
  return Bpk + ((size_t)blockStart << 9) + (lane << 4);
}
DEV v16bf load_b_at(const bf16_t* __restrict__ base, int kb) {
  const bf16_t* p = base + (kb << 9);
  v8bf lo = *(const v8bf*)(p);
  v8bf hi = *(const v8bf*)(p + 8);
  v16bf b;
#pragma unroll
  for (int i = 0; i < 8; ++i) { b[i] = lo[i]; b[i + 8] = hi[i]; }
  return b;
}

DEV v8f wmma_bf16(v16bf a, v16bf b, v8f c) {
  return __builtin_amdgcn_wmma_f32_16x16x32_bf16(false, a, false, b, (short)0, c,
                                                 false, false);
}

// ---------------------------------------------------------------------------
// Pack a row-major f32 weight (K x N) into WMMA B-fragment layout, value (P)
// and |value| (Q), bf16, zero-padded to (Kp x Np).
// Block (kt,nt) index = nt*(Kp/32)+kt (K-fastest for streaming in the K-loop).
// ---------------------------------------------------------------------------
__global__ void pack_weight_kernel(const float* __restrict__ W,
                                   bf16_t* __restrict__ P, bf16_t* __restrict__ Q,
                                   int K, int N, int Kp, int Np) {
  int idx = blockIdx.x * blockDim.x + threadIdx.x;
  int total = Kp * Np;
  if (idx >= total) return;
  int blk  = idx >> 9;
  int lane = (idx >> 4) & 31;
  int i    = idx & 15;
  int kBlocks = Kp >> 5;
  int nt = blk / kBlocks;
  int kt = blk - nt * kBlocks;
  int k = (kt << 5) + ((lane >> 4) << 3) + (i & 7) + ((i >> 3) << 4);   // lanes 16-31 hold K+8; elems 8-15 hold K+16
  int n = (nt << 4) + (lane & 15);
  float v = (k < K && n < N) ? W[(size_t)k * N + n] : 0.0f;
  P[idx] = f2bf(v);
  Q[idx] = f2bf(fabsf(v));
}

// ---------------------------------------------------------------------------
// Generic interval-linear GEMM:  l = Ac@BP - Ar@BQ + bias,  u = ... + ...
// 1 tile / wave (used for the one-shot, bandwidth-bound feed-forward stages).
// Wave-uniform tile guard keeps EXEC full for WMMA.
// ---------------------------------------------------------------------------
__global__ __launch_bounds__(256) void ilin_gemm_kernel(
    const bf16_t* __restrict__ Ac, const bf16_t* __restrict__ Ar, int lda,
    const bf16_t* __restrict__ BP, const bf16_t* __restrict__ BQ, int kBlocks,
    const float* __restrict__ bias, float biasC,
    float* __restrict__ outL, float* __restrict__ outU, int ldc, int nStore,
    int mTiles, int totalTiles) {
  const int lane = threadIdx.x & 31;
  const int tile = blockIdx.x * 8 + (threadIdx.x >> 5);
  if (tile >= totalTiles) return;
  const int tm = tile % mTiles;
  const int tn = tile / mTiles;
  const int row0 = tm << 4;
  v8f p = {};
  v8f q = {};
  const bf16_t* acb = a_base(Ac, (size_t)lda, row0, lane);
  const bf16_t* arb = a_base(Ar, (size_t)lda, row0, lane);
  const bf16_t* bpb = b_base(BP, tn * kBlocks, lane);
  const bf16_t* bqb = b_base(BQ, tn * kBlocks, lane);
#pragma unroll 2
  for (int kb = 0; kb < kBlocks; ++kb) {
    v16bf ac = load_a_at(acb, kb << 5);
    v16bf ar = load_a_at(arb, kb << 5);
    v16bf bp = load_b_at(bpb, kb);
    v16bf bq = load_b_at(bqb, kb);
    p = wmma_bf16(ac, bp, p);
    q = wmma_bf16(ar, bq, q);
  }
  const int col = (tn << 4) + (lane & 15);
  if (col < nStore) {
    const int rb = row0 + ((lane >> 4) << 3);
    const float b = biasC + (bias ? bias[col] : 0.0f);
#pragma unroll
    for (int v = 0; v < 8; ++v) {
      size_t o = (size_t)(rb + v) * ldc + col;
      outL[o] = p[v] - q[v] + b;
      outU[o] = p[v] + q[v] + b;
    }
  }
}

// ---------------------------------------------------------------------------
// LSTM gate dual-GEMM: gl/gu (128 x 2048) from stacked [x;h] (K=1024).
// 2x2 super-tile (32x32 output) per wave: 8 dual accumulators, 8 WMMAs per
// K-block for 8 fragment loads.  256 super-tiles = 32 blocks x 8 waves.
// ---------------------------------------------------------------------------
__global__ __launch_bounds__(256) void gate_gemm_kernel(
    const bf16_t* __restrict__ Xc, const bf16_t* __restrict__ Xr, int xlda,
    const bf16_t* __restrict__ Hc, const bf16_t* __restrict__ Hr,
    const bf16_t* __restrict__ WxP, const bf16_t* __restrict__ WxQ,
    const bf16_t* __restrict__ WhP, const bf16_t* __restrict__ WhQ,
    const float* __restrict__ bias,
    float* __restrict__ gl, float* __restrict__ gu) {
  const int lane = threadIdx.x & 31;
  const int st = blockIdx.x * 8 + (threadIdx.x >> 5);  // 256 super-tiles
  const int sm = st & 3;        // 4 super-rows of 32
  const int sn = st >> 2;       // 64 super-cols of 32
  const int row0 = sm << 5;
  const int tn0  = sn << 1;     // first 16-col tile
  v8f p00 = {}, q00 = {}, p01 = {}, q01 = {};
  v8f p10 = {}, q10 = {}, p11 = {}, q11 = {};

  // x part (K 0..511)
  {
    const bf16_t* ac0 = a_base(Xc, (size_t)xlda, row0,      lane);
    const bf16_t* ar0 = a_base(Xr, (size_t)xlda, row0,      lane);
    const bf16_t* ac1 = a_base(Xc, (size_t)xlda, row0 + 16, lane);
    const bf16_t* ar1 = a_base(Xr, (size_t)xlda, row0 + 16, lane);
    const bf16_t* bp = b_base(WxP, tn0 << 4, lane);  // tile tn0; tn0+1 at +16 blocks
    const bf16_t* bq = b_base(WxQ, tn0 << 4, lane);
#pragma unroll 4
    for (int kb = 0; kb < 16; ++kb) {
      v16bf a0c = load_a_at(ac0, kb << 5);
      v16bf a0r = load_a_at(ar0, kb << 5);
      v16bf a1c = load_a_at(ac1, kb << 5);
      v16bf a1r = load_a_at(ar1, kb << 5);
      v16bf b0p = load_b_at(bp, kb);
      v16bf b0q = load_b_at(bq, kb);
      v16bf b1p = load_b_at(bp, kb + 16);
      v16bf b1q = load_b_at(bq, kb + 16);
      p00 = wmma_bf16(a0c, b0p, p00); q00 = wmma_bf16(a0r, b0q, q00);
      p01 = wmma_bf16(a0c, b1p, p01); q01 = wmma_bf16(a0r, b1q, q01);
      p10 = wmma_bf16(a1c, b0p, p10); q10 = wmma_bf16(a1r, b0q, q10);
      p11 = wmma_bf16(a1c, b1p, p11); q11 = wmma_bf16(a1r, b1q, q11);
    }
  }
  // h part (K 512..1023)
  {
    const bf16_t* ac0 = a_base(Hc, 512, row0,      lane);
    const bf16_t* ar0 = a_base(Hr, 512, row0,      lane);
    const bf16_t* ac1 = a_base(Hc, 512, row0 + 16, lane);
    const bf16_t* ar1 = a_base(Hr, 512, row0 + 16, lane);
    const bf16_t* bp = b_base(WhP, tn0 << 4, lane);
    const bf16_t* bq = b_base(WhQ, tn0 << 4, lane);
#pragma unroll 4
    for (int kb = 0; kb < 16; ++kb) {
      v16bf a0c = load_a_at(ac0, kb << 5);
      v16bf a0r = load_a_at(ar0, kb << 5);
      v16bf a1c = load_a_at(ac1, kb << 5);
      v16bf a1r = load_a_at(ar1, kb << 5);
      v16bf b0p = load_b_at(bp, kb);
      v16bf b0q = load_b_at(bq, kb);
      v16bf b1p = load_b_at(bp, kb + 16);
      v16bf b1q = load_b_at(bq, kb + 16);
      p00 = wmma_bf16(a0c, b0p, p00); q00 = wmma_bf16(a0r, b0q, q00);
      p01 = wmma_bf16(a0c, b1p, p01); q01 = wmma_bf16(a0r, b1q, q01);
      p10 = wmma_bf16(a1c, b0p, p10); q10 = wmma_bf16(a1r, b0q, q10);
      p11 = wmma_bf16(a1c, b1p, p11); q11 = wmma_bf16(a1r, b1q, q11);
    }
  }

  const int c0 = (tn0 << 4) + (lane & 15);
  const int c1 = c0 + 16;
  const int rb = row0 + ((lane >> 4) << 3);
  const float bias0 = bias[c0];
  const float bias1 = bias[c1];
#pragma unroll
  for (int v = 0; v < 8; ++v) {
    int oA = (rb + v) * 2048;        // rows row0 .. row0+15 half
    int oB = (rb + 16 + v) * 2048;   // rows row0+16 .. row0+31 half
    gl[oA + c0] = p00[v] - q00[v] + bias0;
    gu[oA + c0] = p00[v] + q00[v] + bias0;
    gl[oA + c1] = p01[v] - q01[v] + bias1;
    gu[oA + c1] = p01[v] + q01[v] + bias1;
    gl[oB + c0] = p10[v] - q10[v] + bias0;
    gu[oB + c0] = p10[v] + q10[v] + bias0;
    gl[oB + c1] = p11[v] - q11[v] + bias1;
    gu[oB + c1] = p11[v] + q11[v] + bias1;
  }
}

// ---------------------------------------------------------------------------
// Elementwise stages
// ---------------------------------------------------------------------------

// Frame the padded input: m = b*320+f, k in [0,256). pos=f*200+k; >=64000 -> 0.
__global__ void frame_kernel(const float* __restrict__ lb, const float* __restrict__ ub,
                             bf16_t* __restrict__ C, bf16_t* __restrict__ R) {
  int idx = blockIdx.x * 256 + threadIdx.x;  // 40960*256 threads
  int m = idx >> 8;
  int k = idx & 255;
  int b = m / 320;
  int f = m - b * 320;
  int pos = f * 200 + k;
  float l = 0.0f, u = 0.0f;
  if (pos < 64000) {
    size_t o = (size_t)b * 64000 + pos;
    l = lb[o]; u = ub[o];
  }
  C[idx] = f2bf(0.5f * (l + u));
  R[idx] = f2bf(0.5f * (u - l));
}

// Interval square -> center/radius bf16.  (40960 x 288)
__global__ void square_kernel(const float* __restrict__ L, const float* __restrict__ U,
                              bf16_t* __restrict__ C, bf16_t* __restrict__ R) {
  size_t o = (size_t)blockIdx.x * 288 + threadIdx.x;
  float l = L[o], u = U[o];
  float a = l * l, bb = u * u;
  float hi = fmaxf(a, bb);
  float lo = (l <= 0.0f && u >= 0.0f) ? 0.0f : fminf(a, bb);
  C[o] = f2bf(0.5f * (lo + hi));
  R[o] = f2bf(0.5f * (hi - lo));
}

// log -> center/radius bf16. (40960 x 64)
__global__ void log_kernel(const float* __restrict__ L, const float* __restrict__ U,
                           bf16_t* __restrict__ C, bf16_t* __restrict__ R) {
  int idx = blockIdx.x * 256 + threadIdx.x;
  float l = __logf(L[idx]);
  float u = __logf(U[idx]);
  C[idx] = f2bf(0.5f * (l + u));
  R[idx] = f2bf(0.5f * (u - l));
}

// relu -> center/radius bf16. (40960 x 512)
__global__ void relu_kernel(const float* __restrict__ L, const float* __restrict__ U,
                            bf16_t* __restrict__ C, bf16_t* __restrict__ R) {
  int idx = blockIdx.x * 256 + threadIdx.x;
  float l = fmaxf(L[idx], 0.0f);
  float u = fmaxf(U[idx], 0.0f);
  C[idx] = f2bf(0.5f * (l + u));
  R[idx] = f2bf(0.5f * (u - l));
}

// Interval LSTM cell pointwise: reads gl/gu (128 x 2048), updates c (f32 interval)
// and h as center/radius bf16 (consumed by the next gate GEMMs).
__global__ void lstm_cell_kernel(const float* __restrict__ gl, const float* __restrict__ gu,
                                 float* __restrict__ cL, float* __restrict__ cU,
                                 bf16_t* __restrict__ hC, bf16_t* __restrict__ hR) {
  int idx = blockIdx.x * 256 + threadIdx.x;  // 128*512
  int b = idx >> 9;
  int j = idx & 511;
  const float* gLo = gl + b * 2048;
  const float* gHi = gu + b * 2048;
  float il = sigf(gLo[j]),          iu = sigf(gHi[j]);
  float fl = sigf(gLo[512 + j]),    fu = sigf(gHi[512 + j]);
  float tl = tanhf(gLo[1024 + j]),  tu = tanhf(gHi[1024 + j]);
  float ol = sigf(gLo[1536 + j]),   ou = sigf(gHi[1536 + j]);
  float cl = cL[idx], cu = cU[idx];
  float fcl, fcu; imul(fl, fu, cl, cu, fcl, fcu);
  float igl, igu; imul(il, iu, tl, tu, igl, igu);
  float ncl = fcl + igl, ncu = fcu + igu;
  float thl = tanhf(ncl), thu = tanhf(ncu);
  float nhl, nhu; imul(ol, ou, thl, thu, nhl, nhu);
  cL[idx] = ncl;
  cU[idx] = ncu;
  hC[idx] = f2bf(0.5f * (nhl + nhu));
  hR[idx] = f2bf(0.5f * (nhu - nhl));
}

// ---------------------------------------------------------------------------
// Host orchestration
// ---------------------------------------------------------------------------
extern "C" void kernel_launch(void* const* d_in, const int* in_sizes, int n_in,
                              void* d_out, int out_size, void* d_ws, size_t ws_size,
                              hipStream_t stream) {
  (void)in_sizes; (void)n_in; (void)out_size;
  const float* input_lb = (const float*)d_in[0];
  const float* input_ub = (const float*)d_in[1];
  const float* mtx1     = (const float*)d_in[2];
  const float* mtx2     = (const float*)d_in[3];
  const float* lin1_w   = (const float*)d_in[4];
  const float* lin1_b   = (const float*)d_in[5];
  const float* w_ih0    = (const float*)d_in[6];
  const float* w_hh0    = (const float*)d_in[7];
  const float* b0       = (const float*)d_in[8];
  const float* w_ih1    = (const float*)d_in[9];
  const float* w_hh1    = (const float*)d_in[10];
  const float* b1       = (const float*)d_in[11];
  const float* lin2_w   = (const float*)d_in[12];
  const float* lin2_b   = (const float*)d_in[13];

  const int M = 40960;  // B*F = 128*320

  size_t off = 0;
  char* base = (char*)d_ws;
  auto alloc = [&](size_t bytes) -> char* {
    char* p = base + off;
    off += (bytes + 255) & ~(size_t)255;
    return p;
  };

  // Packed weights (bf16, fragment-major).  P = value, Q = |value|.
  bf16_t* pk1P  = (bf16_t*)alloc((size_t)256 * 288 * 2);
  bf16_t* pk1Q  = (bf16_t*)alloc((size_t)256 * 288 * 2);
  bf16_t* pk2P  = (bf16_t*)alloc((size_t)288 * 64 * 2);
  bf16_t* pk2Q  = (bf16_t*)alloc((size_t)288 * 64 * 2);
  bf16_t* pkL1P = (bf16_t*)alloc((size_t)64 * 512 * 2);
  bf16_t* pkL1Q = (bf16_t*)alloc((size_t)64 * 512 * 2);
  bf16_t* pkL2P = (bf16_t*)alloc((size_t)512 * 16 * 2);
  bf16_t* pkL2Q = (bf16_t*)alloc((size_t)512 * 16 * 2);
  const size_t WSZ = (size_t)512 * 2048;
  bf16_t* pkIh0P = (bf16_t*)alloc(WSZ * 2); bf16_t* pkIh0Q = (bf16_t*)alloc(WSZ * 2);
  bf16_t* pkHh0P = (bf16_t*)alloc(WSZ * 2); bf16_t* pkHh0Q = (bf16_t*)alloc(WSZ * 2);
  bf16_t* pkIh1P = (bf16_t*)alloc(WSZ * 2); bf16_t* pkIh1Q = (bf16_t*)alloc(WSZ * 2);
  bf16_t* pkHh1P = (bf16_t*)alloc(WSZ * 2); bf16_t* pkHh1Q = (bf16_t*)alloc(WSZ * 2);

  // Activations
  bf16_t* frC = (bf16_t*)alloc((size_t)M * 256 * 2);
  bf16_t* frR = (bf16_t*)alloc((size_t)M * 256 * 2);
  float*  s1L = (float*)alloc((size_t)M * 288 * 4);
  float*  s1U = (float*)alloc((size_t)M * 288 * 4);
  bf16_t* sqC = (bf16_t*)alloc((size_t)M * 288 * 2);
  bf16_t* sqR = (bf16_t*)alloc((size_t)M * 288 * 2);
  float*  s2L = (float*)alloc((size_t)M * 64 * 4);
  float*  s2U = (float*)alloc((size_t)M * 64 * 4);
  bf16_t* lgC = (bf16_t*)alloc((size_t)M * 64 * 2);
  bf16_t* lgR = (bf16_t*)alloc((size_t)M * 64 * 2);
  float*  n1L = (float*)alloc((size_t)M * 512 * 4);
  float*  n1U = (float*)alloc((size_t)M * 512 * 4);
  bf16_t* xC  = (bf16_t*)alloc((size_t)M * 512 * 2);
  bf16_t* xR  = (bf16_t*)alloc((size_t)M * 512 * 2);
  float*  gl  = (float*)alloc((size_t)128 * 2048 * 4);
  float*  gu  = (float*)alloc((size_t)128 * 2048 * 4);
  float*  c0L = (float*)alloc((size_t)65536 * 4);
  float*  c0U = (float*)alloc((size_t)65536 * 4);
  float*  c1L = (float*)alloc((size_t)65536 * 4);
  float*  c1U = (float*)alloc((size_t)65536 * 4);
  bf16_t* h0C = (bf16_t*)alloc((size_t)65536 * 2);
  bf16_t* h0R = (bf16_t*)alloc((size_t)65536 * 2);
  bf16_t* h1C = (bf16_t*)alloc((size_t)65536 * 2);
  bf16_t* h1R = (bf16_t*)alloc((size_t)65536 * 2);

  if (off > ws_size) return;  // workspace too small; nothing safe to do

  auto pack = [&](const float* W, bf16_t* P, bf16_t* Q, int K, int N, int Kp, int Np) {
    int total = Kp * Np;
    pack_weight_kernel<<<(total + 255) / 256, 256, 0, stream>>>(W, P, Q, K, N, Kp, Np);
  };
  pack(mtx1,   pk1P,  pk1Q,  256, 258, 256, 288);
  pack(mtx2,   pk2P,  pk2Q,  258, 64,  288, 64);
  pack(lin1_w, pkL1P, pkL1Q, 64,  512, 64,  512);
  pack(lin2_w, pkL2P, pkL2Q, 512, 10,  512, 16);
  pack(w_ih0,  pkIh0P, pkIh0Q, 512, 2048, 512, 2048);
  pack(w_hh0,  pkHh0P, pkHh0Q, 512, 2048, 512, 2048);
  pack(w_ih1,  pkIh1P, pkIh1Q, 512, 2048, 512, 2048);
  pack(w_hh1,  pkHh1P, pkHh1Q, 512, 2048, 512, 2048);

  // Feed-forward stages
  frame_kernel<<<40960, 256, 0, stream>>>(input_lb, input_ub, frC, frR);

  // stage1: M=40960, N=288(pad of 258), K=256 -> 2560*18 tiles
  ilin_gemm_kernel<<<(2560 * 18 + 7) / 8, 256, 0, stream>>>(
      frC, frR, 256, pk1P, pk1Q, 8, nullptr, 0.0f, s1L, s1U, 288, 288, 2560, 2560 * 18);

  square_kernel<<<40960, 288, 0, stream>>>(s1L, s1U, sqC, sqR);

  // stage2: N=64, K=288 -> 2560*4 tiles
  ilin_gemm_kernel<<<(2560 * 4 + 7) / 8, 256, 0, stream>>>(
      sqC, sqR, 288, pk2P, pk2Q, 9, nullptr, 1e-10f, s2L, s2U, 64, 64, 2560, 2560 * 4);

  log_kernel<<<(M * 64) / 256, 256, 0, stream>>>(s2L, s2U, lgC, lgR);

  // lin1: N=512, K=64 -> 2560*32 tiles
  ilin_gemm_kernel<<<(2560 * 32 + 7) / 8, 256, 0, stream>>>(
      lgC, lgR, 64, pkL1P, pkL1Q, 2, lin1_b, 0.0f, n1L, n1U, 512, 512, 2560, 2560 * 32);

  relu_kernel<<<(M * 512) / 256, 256, 0, stream>>>(n1L, n1U, xC, xR);

  // Zero LSTM state
  hipMemsetAsync(c0L, 0, 65536 * 4, stream);
  hipMemsetAsync(c0U, 0, 65536 * 4, stream);
  hipMemsetAsync(c1L, 0, 65536 * 4, stream);
  hipMemsetAsync(c1U, 0, 65536 * 4, stream);
  hipMemsetAsync(h0C, 0, 65536 * 2, stream);
  hipMemsetAsync(h0R, 0, 65536 * 2, stream);
  hipMemsetAsync(h1C, 0, 65536 * 2, stream);
  hipMemsetAsync(h1R, 0, 65536 * 2, stream);

  // Sequential LSTM: layer0 x-rows live at (b*320+t)*512 -> lda=320*512, base +t*512
  for (int t = 0; t < 320; ++t) {
    gate_gemm_kernel<<<32, 256, 0, stream>>>(
        xC + (size_t)t * 512, xR + (size_t)t * 512, 320 * 512,
        h0C, h0R, pkIh0P, pkIh0Q, pkHh0P, pkHh0Q, b0, gl, gu);
    lstm_cell_kernel<<<256, 256, 0, stream>>>(gl, gu, c0L, c0U, h0C, h0R);
    gate_gemm_kernel<<<32, 256, 0, stream>>>(
        h0C, h0R, 512,
        h1C, h1R, pkIh1P, pkIh1Q, pkHh1P, pkHh1Q, b1, gl, gu);
    lstm_cell_kernel<<<256, 256, 0, stream>>>(gl, gu, c1L, c1U, h1C, h1R);
  }

  // Final lin2: M=128, N=16(pad of 10), K=512.  out = stack([ol, ou]) (2,128,10)
  float* outL = (float*)d_out;
  float* outU = (float*)d_out + 128 * 10;
  ilin_gemm_kernel<<<1, 256, 0, stream>>>(
      h1C, h1R, 512, pkL2P, pkL2Q, 16, lin2_b, 0.0f, outL, outU, 10, 10, 8, 8);
}
